// BinaryLinear_51719996178701
// MI455X (gfx1250) — 8-bit matrix operand order repaired and checked by reading, compiled, not run on hardware
//
#include <hip/hip_runtime.h>
#include <hip/hip_bf16.h>
#include <stdint.h>

#define M_DIM 16384
#define K_DIM 2048
#define N_DIM 2048

#define FP8_P1 0x38u
#define FP8_N1 0xB8u

typedef __attribute__((ext_vector_type(16))) int   v16i;
typedef __attribute__((ext_vector_type(8)))  float v8f;
typedef __attribute__((ext_vector_type(4)))  int   v4i;

typedef __attribute__((address_space(1))) v4i glb_v4i;
typedef __attribute__((address_space(3))) v4i lds_v4i;

__device__ __forceinline__ void async_cp_b128(const uint8_t* g, uint8_t* l) {
#if __has_builtin(__builtin_amdgcn_global_load_async_to_lds_b128)
  __builtin_amdgcn_global_load_async_to_lds_b128(
      (glb_v4i*)(uintptr_t)g,
      (lds_v4i*)(uint32_t)(uintptr_t)l,
       0,  0);
#else
  uint32_t lds_off = (uint32_t)(uintptr_t)l;
  uint64_t gaddr   = (uint64_t)(uintptr_t)g;
  asm volatile("global_load_async_to_lds_b128 %0, %1, off"
               :: "v"(lds_off), "v"(gaddr)
               : "memory");
#endif
}

__device__ __forceinline__ void wait_asynccnt0() {
#if __has_builtin(__builtin_amdgcn_s_wait_asynccnt)
  __builtin_amdgcn_s_wait_asynccnt(0);
#else
  asm volatile("s_wait_asynccnt 0x0" ::: "memory");
#endif
}

__global__ void quantize_sign_fp8(const float* __restrict__ in,
                                  uint32_t* __restrict__ outq, int n4) {
  int i = blockIdx.x * blockDim.x + threadIdx.x;
  if (i >= n4) return;
  float4 v = ((const float4*)in)[i];
  uint32_t b0 = (v.x < 0.0f) ? FP8_N1 : FP8_P1;
  uint32_t b1 = (v.y < 0.0f) ? FP8_N1 : FP8_P1;
  uint32_t b2 = (v.z < 0.0f) ? FP8_N1 : FP8_P1;
  uint32_t b3 = (v.w < 0.0f) ? FP8_N1 : FP8_P1;
  outq[i] = b0 | (b1 << 8) | (b2 << 16) | (b3 << 24);
}

#define BM 128
#define BN 128
#define BK 128
#define LDSS 144

__global__ __launch_bounds__(256)
void bnn_gemm_fp8(const uint8_t* __restrict__ xq,
                  const uint8_t* __restrict__ wq,
                  float* __restrict__ out) {
  __shared__ uint8_t ldsA[2][BM * LDSS];
  __shared__ uint8_t ldsB[2][BN * LDSS];

  const int t     = threadIdx.x;
  const int lane  = t & 31;
  const int wv    = t >> 5;
  const int waveM = wv >> 2;
  const int waveN = wv & 3;
  const int m0    = blockIdx.y * BM;
  const int n0    = blockIdx.x * BN;

  const int cpRow   = t >> 3;
  const int cpChunk = (t & 7) * 16;

  v8f acc[4][2];
#pragma unroll
  for (int i = 0; i < 4; ++i)
#pragma unroll
    for (int j = 0; j < 2; ++j)
      acc[i][j] = (v8f){0.f, 0.f, 0.f, 0.f, 0.f, 0.f, 0.f, 0.f};

  const int NK = K_DIM / BK;

#pragma unroll
  for (int r = 0; r < 4; ++r) {
    int row = cpRow + r * 32;
    async_cp_b128(&xq[(size_t)(m0 + row) * K_DIM + cpChunk],
                  &ldsA[0][row * LDSS + cpChunk]);
    async_cp_b128(&wq[(size_t)(n0 + row) * K_DIM + cpChunk],
                  &ldsB[0][row * LDSS + cpChunk]);
  }
  wait_asynccnt0();
  __syncthreads();

  const int laneLo = lane & 15;
  const int aKoff  = (lane >> 4) * 8;
  const int bKoff  = (lane >> 4) * 16;

  for (int kt = 0; kt < NK; ++kt) {
    const int s = kt & 1;

    if (kt + 1 < NK) {
      const int kb = (kt + 1) * BK;
#pragma unroll
      for (int r = 0; r < 4; ++r) {
        int row = cpRow + r * 32;
        async_cp_b128(&xq[(size_t)(m0 + row) * K_DIM + kb + cpChunk],
                      &ldsA[s ^ 1][row * LDSS + cpChunk]);
        async_cp_b128(&wq[(size_t)(n0 + row) * K_DIM + kb + cpChunk],
                      &ldsB[s ^ 1][row * LDSS + cpChunk]);
      }
    }

    v16i bfrag[2];
#pragma unroll
    for (int nt = 0; nt < 2; ++nt) {
      const uint8_t* brow =
          &ldsB[s][(waveN * 32 + nt * 16 + laneLo) * LDSS];
#pragma unroll
      for (int j = 0; j < 4; ++j) {
        v4i q = *(const v4i*)(brow + 32 * j + bKoff);
        bfrag[nt][4 * j + 0] = q[0];
        bfrag[nt][4 * j + 1] = q[1];
        bfrag[nt][4 * j + 2] = q[2];
        bfrag[nt][4 * j + 3] = q[3];
      }
    }

#pragma unroll
    for (int mt = 0; mt < 4; ++mt) {
      const uint8_t* arow =
          &ldsA[s][(waveM * 64 + mt * 16 + laneLo) * LDSS];
      v16i afrag;
#pragma unroll
      for (int h = 0; h < 2; ++h) {
#pragma unroll
        for (int q = 0; q < 4; ++q) {
          uint64_t d = *(const uint64_t*)(arow + 64 * h + 32 * (q >> 1) + 8 * (q & 1) + 2 * aKoff);
          afrag[8 * h + 2 * q + 0] = (int)(uint32_t)d;
          afrag[8 * h + 2 * q + 1] = (int)(uint32_t)(d >> 32);
        }
      }
#pragma unroll
      for (int nt = 0; nt < 2; ++nt) {
        acc[mt][nt] = __builtin_amdgcn_wmma_f32_16x16x128_fp8_fp8(
            afrag, bfrag[nt], (short)0, acc[mt][nt],
             false,  false);
      }
    }

    wait_asynccnt0();
    __syncthreads();
  }

  const int mAdd = (lane >> 4) * 8;
#pragma unroll
  for (int mt = 0; mt < 4; ++mt) {
#pragma unroll
    for (int nt = 0; nt < 2; ++nt) {
      const int n = n0 + waveN * 32 + nt * 16 + laneLo;
#pragma unroll
      for (int j = 0; j < 8; ++j) {
        const int m = m0 + waveM * 64 + mt * 16 + j + mAdd;
        out[(size_t)m * N_DIM + n] = acc[mt][nt][j];
      }
    }
  }
}

extern "C" void kernel_launch(void* const* d_in, const int* in_sizes, int n_in,
                              void* d_out, int out_size, void* d_ws, size_t ws_size,
                              hipStream_t stream) {
  const float* x = (const float*)d_in[0];
  const float* w = (const float*)d_in[1];
  float* out = (float*)d_out;

  uint8_t* xq = (uint8_t*)d_ws;
  uint8_t* wq = (uint8_t*)d_ws + (size_t)M_DIM * K_DIM;

  {
    int n4 = (M_DIM * K_DIM) / 4;
    quantize_sign_fp8<<<n4 / 256, 256, 0, stream>>>(x, (uint32_t*)xq, n4);
  }
  {
    int n4 = (N_DIM * K_DIM) / 4;
    quantize_sign_fp8<<<n4 / 256, 256, 0, stream>>>(w, (uint32_t*)wq, n4);
  }

  dim3 grid(N_DIM / BN, M_DIM / BM);
  bnn_gemm_fp8<<<grid, 256, 0, stream>>>(xq, wq, out);
}
